// MovieRecommendationModel_19241453486451
// MI455X (gfx1250) — compile-verified, matrix operand order checked by reading, not run on hardware
//
#include <hip/hip_runtime.h>
#include <hip/hip_bf16.h>
#include <math.h>

typedef __attribute__((ext_vector_type(16))) _Float16 v16h;
typedef __attribute__((ext_vector_type(8)))  _Float16 v8h;
typedef __attribute__((ext_vector_type(8)))  float    v8f;

// ---------------------------------------------------------------------------
// Input indices (setup_inputs() dict order, params dict flattened in order)
// ---------------------------------------------------------------------------
enum {
  IN_USER_ID = 0, IN_MOVIE_ID, IN_GENDER, IN_AGE, IN_JOB,
  IN_CATS, IN_TITLES, IN_POSTER,
  IN_USER_EMB, IN_GENDER_EMB, IN_AGE_EMB, IN_JOB_EMB, IN_MOVIE_EMB,
  IN_CAT_EMB, IN_TITLE_EMB,
  IN_C1W, IN_C1B, IN_C2W, IN_C2B, IN_C3W, IN_C3B, IN_PFW, IN_PFB,
  IN_T2W, IN_T2B, IN_T3W, IN_T3B, IN_T4W, IN_T4B, IN_T5W, IN_T5B,
  IN_TFW, IN_TFB,
  IN_WQ, IN_BQ, IN_WK, IN_BK, IN_WV, IN_BV, IN_WO, IN_BO,
  IN_F1W, IN_F1B, IN_F2W, IN_F2B,
  IN_UW, IN_UB, IN_GW, IN_GB, IN_AW, IN_AB, IN_JW, IN_JB,
  IN_UFW, IN_UFB, IN_MFW, IN_MFB,
  IN_COUNT
};

#define BATCH 16384

// workspace layout: [0, 2MB) poster_features f32 [B,32]; then f16 fragments
#define WS_FRAG_BYTES ((size_t)BATCH * 32 * sizeof(float))
// fragment region offsets (in halves)
enum { FR_C1 = 0, FR_C2 = 512, FR_DENSE = 512 + 10 * 512,
       N_DENSE_FRAGS = 28,
       FR_TOTAL = 512 + 10 * 512 + N_DENSE_FRAGS * 512 };

// dense fragment bases (order inside a layer: nt0 ks0..NK-1, nt1 ks0..NK-1)
enum { FB_WV = 0, FB_WO = 2, FB_F1 = 4, FB_F2 = 8, FB_MF = 10,
       FB_UW = 12, FB_GW = 14, FB_AW = 16, FB_JW = 18, FB_UF = 20 };

// ---------------------------------------------------------------------------
// WMMA fragment index helper (CDNA5 16-bit A/B layout, wave32):
//   lanes 0-15 : row/col = lane,    K elements {0..7, 16..23}
//   lanes 16-31: row/col = lane-16, K elements {8..15, 24..31}
// => per lane the fragment is TWO contiguous 8-half (16B) runs.
// ---------------------------------------------------------------------------
__device__ __forceinline__ int a_k(int lane, int i) {
  int base = (lane & 16) ? 8 : 0;
  return (i < 8) ? (i + base) : (8 + i + base);   // i>=8 -> 16 + (i-8) + base
}

__device__ __forceinline__ v8f wmma_f16(v16h a, v16h b, v8f c) {
  return __builtin_amdgcn_wmma_f32_16x16x32_f16(false, a, false, b,
                                                (short)0, c, false, false);
}

__device__ __forceinline__ v8h zero_v8h() {
  v8h z;
#pragma unroll
  for (int i = 0; i < 8; ++i) z[i] = (_Float16)0.f;
  return z;
}

__device__ __forceinline__ v16h combine16(v8h lo, v8h hi) {
  v16h r;
#pragma unroll
  for (int i = 0; i < 8; ++i) { r[i] = lo[i]; r[8 + i] = hi[i]; }
  return r;
}

// A-fragment from an LDS row of 32 contiguous K halves: two 16B ds_load_b128.
__device__ __forceinline__ v16h afrag_row(const _Float16* row, int lane) {
  const int base = (lane & 16) ? 8 : 0;
  v8h lo = *(const v8h*)(row + base);
  v8h hi = *(const v8h*)(row + 16 + base);
  return combine16(lo, hi);
}

__device__ __forceinline__ v16h afrag_row_k16(const _Float16* row, int lane) {
  const int base = (lane & 16) ? 8 : 0;     // K<=16: hi half is zero padding
  v8h lo = *(const v8h*)(row + base);
  return combine16(lo, zero_v8h());
}

// ---------------------------------------------------------------------------
// Dense-layer weight pointers (attention query path is numerically dead:
// softmax over kv-len 1 == 1.0 exactly, ctx == v bit-exactly, so the whole
// movie/title/category branch is eliminated).
// ---------------------------------------------------------------------------
struct DenseP {
  const int *uid, *gid, *aid, *jid;
  const float *ue, *ge, *ae, *je;  // embedding tables
  const float *wv, *bv, *wo, *bo;
  const float *f1w, *f1b, *f2w, *f2b;
  const float *uw, *ub, *gw, *gb, *aw, *ab, *jw, *jb;
  const float *ufw, *ufb, *mfw, *mfb;
};

// ---------------------------------------------------------------------------
// Kernel 0: one-block init — pre-swizzle every weight matrix into the exact
// per-lane WMMA B-fragment layout ([frag][lane][16] halves) in workspace.
// Launch-invariant; compute kernels then fetch each fragment with a single
// 32-byte global_load_b128 from L2.
// ---------------------------------------------------------------------------
__global__ __launch_bounds__(256) void init_frags_kernel(
    const float* __restrict__ c1w, const float* __restrict__ c2w,
    DenseP P, _Float16* __restrict__ fr)
{
  const int tid = threadIdx.x;
  // conv1: K=27 pad 32, N=16
  for (int e = tid; e < 512; e += 256) {
    int ln = e >> 4, i = e & 15;
    int k = a_k(ln, i), n = ln & 15;
    float v = 0.f;
    if (k < 27) {
      int c = k / 9, r = (k % 9) / 3, s = k % 3;
      v = c1w[((n * 3 + c) * 3 + r) * 3 + s];
    }
    fr[FR_C1 + e] = (_Float16)v;
  }
  // conv2: 10 frags (nt*5+ks), K=144 pad 160, N=32
  for (int e = tid; e < 10 * 512; e += 256) {
    int f = e >> 9, r9 = e & 511;
    int ln = r9 >> 4, i = r9 & 15;
    int k = (f % 5) * 32 + a_k(ln, i);
    int n = (f / 5) * 16 + (ln & 15);
    float v = 0.f;
    if (k < 144) {
      int c = k / 9, r = (k % 9) / 3, s = k % 3;
      v = c2w[((n * 16 + c) * 3 + r) * 3 + s];
    }
    fr[FR_C2 + e] = (_Float16)v;
  }
  // dense: 28 frags
  struct FragD { const float* W; int Kin, ks, nt; };
  const FragD fds[N_DENSE_FRAGS] = {
    {P.wv, 32, 0, 0},  {P.wv, 32, 0, 1},
    {P.wo, 32, 0, 0},  {P.wo, 32, 0, 1},
    {P.f1w, 64, 0, 0}, {P.f1w, 64, 1, 0}, {P.f1w, 64, 0, 1}, {P.f1w, 64, 1, 1},
    {P.f2w, 32, 0, 0}, {P.f2w, 32, 0, 1},
    {P.mfw, 32, 0, 0}, {P.mfw, 32, 0, 1},
    {P.uw, 32, 0, 0},  {P.uw, 32, 0, 1},
    {P.gw, 16, 0, 0},  {P.gw, 16, 0, 1},
    {P.aw, 16, 0, 0},  {P.aw, 16, 0, 1},
    {P.jw, 16, 0, 0},  {P.jw, 16, 0, 1},
    {P.ufw, 128, 0, 0}, {P.ufw, 128, 1, 0}, {P.ufw, 128, 2, 0}, {P.ufw, 128, 3, 0},
    {P.ufw, 128, 0, 1}, {P.ufw, 128, 1, 1}, {P.ufw, 128, 2, 1}, {P.ufw, 128, 3, 1},
  };
#pragma unroll
  for (int f = 0; f < N_DENSE_FRAGS; ++f) {
    const float* W = fds[f].W;
    const int Kin = fds[f].Kin, ks = fds[f].ks, nt = fds[f].nt;
    for (int e = tid; e < 512; e += 256) {
      int ln = e >> 4, i = e & 15;
      int k = ks * 32 + a_k(ln, i);
      int n = nt * 16 + (ln & 15);
      fr[FR_DENSE + f * 512 + e] =
          (k < Kin) ? (_Float16)W[n * Kin + k] : (_Float16)0.f;
    }
  }
}

// ---------------------------------------------------------------------------
// Kernel 1: poster CNN. One workgroup (8 waves, 256 thr) per sample.
//   conv1: 3->16ch 3x3 s2 p1, 32x32->16x16  : WMMA (im2col in LDS, K=27->32)
//   pool -> 8x8 ; conv2: 16->32ch -> 4x4    : WMMA (im2col in LDS, K=144->160)
//   pool -> 2x2 ; conv3: 32->64ch -> 1x1 + poster_fc : VALU (tiny)
// All staging index math is shift/mask only; weight fragments come
// pre-swizzled from global (one global_load_b128 each).
// ---------------------------------------------------------------------------
__global__ __launch_bounds__(256) void poster_cnn_kernel(
    const float* __restrict__ img,                       // [B,3,32,32]
    const _Float16* __restrict__ fr,                     // fragment region
    const float* __restrict__ c1b, const float* __restrict__ c2b,
    const float* __restrict__ c3w, const float* __restrict__ c3b,
    const float* __restrict__ pfw, const float* __restrict__ pfb,
    float* __restrict__ pf_out)                          // [B,32]
{
  __shared__ __align__(16) _Float16 sh_img[3][34][34];   // padded image
  __shared__ __align__(16) _Float16 sh_ic[256][32];      // conv1 im2col
  __shared__ __align__(16) _Float16 sh_c1o[16][16][16];  // conv1 out [c][y][x]
  __shared__ __align__(16) _Float16 sh_p1[16][10][10];   // pooled + padded
  __shared__ __align__(16) _Float16 sh_ic2[16][160];     // conv2 im2col
  __shared__ __align__(16) _Float16 sh_c2o[32][4][4];    // conv2 out
  __shared__ __align__(16) _Float16 sh_c3i[32][4][4];    // pooled + padded
  __shared__ float sh_c3o[64];

  const int b    = blockIdx.x;
  const int tid  = threadIdx.x;
  const int lane = tid & 31;
  const int wave = tid >> 5;
  const float* imgb = img + (size_t)b * 3 * 32 * 32;

  // ---- prefetch next sample's image into L2 (global_prefetch_b8) ----
  if (b + 1 < BATCH && tid < 32) {
    const char* nxt = (const char*)(img + (size_t)(b + 1) * 3 * 32 * 32);
    __builtin_prefetch(nxt + tid * 384, 0, 1);
  }

  // ---- phase 0: zero the padded buffers (u32 stores; borders stay 0) ----
  {
    unsigned* zi = (unsigned*)&sh_img[0][0][0];   // 1734 words
    unsigned* zp = (unsigned*)&sh_p1[0][0][0];    // 800 words
    unsigned* zc = (unsigned*)&sh_c3i[0][0][0];   // 256 words
    for (int i = tid; i < 1734; i += 256) zi[i] = 0u;
    for (int i = tid; i < 800;  i += 256) zp[i] = 0u;
    for (int i = tid; i < 256;  i += 256) zc[i] = 0u;
  }
  __syncthreads();

  // ---- stage image interior: linear coalesced reads, shift-only scatter ----
  for (int idx = tid; idx < 3072; idx += 256) {
    int c = idx >> 10, y = (idx >> 5) & 31, x = idx & 31;
    sh_img[c][y + 1][x + 1] = (_Float16)imgb[idx];
  }
  __syncthreads();

  // ---- conv1 im2col: one thread per output position ----
  {
    const int p = tid, oy = p >> 4, ox = p & 15;
    _Float16* row = &sh_ic[p][0];
#pragma unroll
    for (int c = 0; c < 3; ++c)
#pragma unroll
      for (int r = 0; r < 3; ++r)
#pragma unroll
        for (int s = 0; s < 3; ++s)
          row[c * 9 + r * 3 + s] = sh_img[c][2 * oy + r][2 * ox + s];
#pragma unroll
    for (int k = 27; k < 32; ++k) row[k] = (_Float16)0.f;
  }
  __syncthreads();

  // ---- conv1 WMMA: 16 M-tiles (output rows), N=16 channels ----
  {
    const int n  = lane & 15;
    const int m  = lane & 15;
    const int mb = (lane & 16) ? 8 : 0;
    const v16h bfv = *(const v16h*)(fr + FR_C1 + lane * 16);
    const float bias = c1b[n];
    for (int t = wave; t < 16; t += 8) {      // oy = t, ox = m
      v16h af = afrag_row(&sh_ic[t * 16 + m][0], lane);
      v8f acc = {0.f, 0.f, 0.f, 0.f, 0.f, 0.f, 0.f, 0.f};
      acc = wmma_f16(af, bfv, acc);
#pragma unroll
      for (int j = 0; j < 8; ++j)
        sh_c1o[n][t][mb + j] = (_Float16)fmaxf(acc[j] + bias, 0.f);
    }
  }
  __syncthreads();

  // ---- maxpool 16x16 -> 8x8 (interior of padded sh_p1) ----
  for (int idx = tid; idx < 1024; idx += 256) {
    int c = idx >> 6, y = (idx >> 3) & 7, x = idx & 7;
    float a0 = (float)sh_c1o[c][2 * y][2 * x];
    float a1 = (float)sh_c1o[c][2 * y][2 * x + 1];
    float a2 = (float)sh_c1o[c][2 * y + 1][2 * x];
    float a3 = (float)sh_c1o[c][2 * y + 1][2 * x + 1];
    sh_p1[c][y + 1][x + 1] = (_Float16)fmaxf(fmaxf(a0, a1), fmaxf(a2, a3));
  }
  __syncthreads();

  // ---- conv2 im2col: thread = (channel, position); 9 contiguous K-slots ----
  {
    const int c = tid >> 4, p = tid & 15;
    const int oy = p >> 2, ox = p & 3;
#pragma unroll
    for (int r = 0; r < 3; ++r)
#pragma unroll
      for (int s = 0; s < 3; ++s)
        sh_ic2[p][c * 9 + r * 3 + s] = sh_p1[c][2 * oy + r][2 * ox + s];
    if (c == 0) {
#pragma unroll
      for (int k = 144; k < 160; ++k) sh_ic2[p][k] = (_Float16)0.f;
    }
  }
  __syncthreads();

  // ---- conv2 WMMA: M=16 positions, 2 N-tiles (waves 0/1), 5 K-steps ----
  if (wave < 2) {
    const int nt = wave;
    const int n  = lane & 15;
    const int m  = lane & 15;
    v8f acc = {0.f, 0.f, 0.f, 0.f, 0.f, 0.f, 0.f, 0.f};
#pragma unroll
    for (int ks = 0; ks < 5; ++ks) {
      v16h af  = afrag_row(&sh_ic2[m][ks * 32], lane);
      v16h bfv = *(const v16h*)(fr + FR_C2 + (nt * 5 + ks) * 512 + lane * 16);
      acc = wmma_f16(af, bfv, acc);
    }
    const int mb = (lane & 16) ? 8 : 0;
    const float bias = c2b[nt * 16 + n];
#pragma unroll
    for (int j = 0; j < 8; ++j) {
      int m2 = mb + j;
      sh_c2o[nt * 16 + n][m2 >> 2][m2 & 3] = (_Float16)fmaxf(acc[j] + bias, 0.f);
    }
  }
  __syncthreads();

  // ---- maxpool 4x4 -> 2x2 (interior of padded sh_c3i) ----
  if (tid < 128) {
    int c = tid >> 2, y = (tid >> 1) & 1, x = tid & 1;
    float a0 = (float)sh_c2o[c][2 * y][2 * x];
    float a1 = (float)sh_c2o[c][2 * y][2 * x + 1];
    float a2 = (float)sh_c2o[c][2 * y + 1][2 * x];
    float a3 = (float)sh_c2o[c][2 * y + 1][2 * x + 1];
    sh_c3i[c][y + 1][x + 1] = (_Float16)fmaxf(fmaxf(a0, a1), fmaxf(a2, a3));
  }
  __syncthreads();

  // ---- conv3 (single output position) + relu + 1x1 avgpool(identity) ----
  if (tid < 64) {
    const int o = tid;
    float acc = c3b[o];
    for (int c = 0; c < 32; ++c)
#pragma unroll
      for (int r = 0; r < 3; ++r)
#pragma unroll
        for (int s = 0; s < 3; ++s)
          acc += (float)sh_c3i[c][r][s] * c3w[((o * 32 + c) * 3 + r) * 3 + s];
    sh_c3o[o] = fmaxf(acc, 0.f);
  }
  __syncthreads();

  // ---- poster_fc: 64 -> 32 ----
  if (tid < 32) {
    const int n = tid;
    float acc = pfb[n];
    for (int i = 0; i < 64; ++i) acc += sh_c3o[i] * pfw[n * 64 + i];
    pf_out[(size_t)b * 32 + n] = acc;
  }
}

// ---------------------------------------------------------------------------
// 16-sample x 32-output linear on WMMA with prestaged global weight frags.
// src/dst: per-wave LDS activation tiles. ACT: 0=id, 1=relu, 2=tanh.
// A-fragments preloaded before any store -> in-place (src==dst) is safe.
// ---------------------------------------------------------------------------
template<int Kin, int ACT>
__device__ __forceinline__ void linear16(const _Float16* src, int ldS, int koff,
                                         const _Float16* __restrict__ wfr,
                                         int fragBase,
                                         const float* __restrict__ bias,
                                         _Float16* dst, int ldD, int doff,
                                         int lane) {
  constexpr int NK = (Kin + 31) / 32;
  const int m = lane & 15;
  v16h afr[NK];
#pragma unroll
  for (int ks = 0; ks < NK; ++ks) {
    const _Float16* row = src + m * ldS + koff + ks * 32;
    if (Kin - ks * 32 >= 32) afr[ks] = afrag_row(row, lane);
    else                     afr[ks] = afrag_row_k16(row, lane);
  }
  const int n  = lane & 15;
  const int mb = (lane & 16) ? 8 : 0;
#pragma unroll
  for (int nt = 0; nt < 2; ++nt) {
    v8f acc = {0.f, 0.f, 0.f, 0.f, 0.f, 0.f, 0.f, 0.f};
#pragma unroll
    for (int ks = 0; ks < NK; ++ks) {
      v16h bf = *(const v16h*)(wfr + ((fragBase + nt * NK + ks) * 32 + lane) * 16);
      acc = wmma_f16(afr[ks], bf, acc);
    }
    float bv = bias[nt * 16 + n];
#pragma unroll
    for (int j = 0; j < 8; ++j) {
      float v = acc[j] + bv;
      if (ACT == 1) v = fmaxf(v, 0.f);
      if (ACT == 2) v = tanhf(v);
      dst[(mb + j) * ldD + doff + nt * 16 + n] = (_Float16)v;
    }
  }
}

// ---------------------------------------------------------------------------
// Kernel 2: dense tower stack. One wave per 16 samples; 8 waves per block.
// No barriers: each wave owns its LDS tiles; weight frags stream from L2.
// ---------------------------------------------------------------------------
__global__ __launch_bounds__(256) void dense_wmma_kernel(
    const float* __restrict__ pf, const _Float16* __restrict__ dfr,
    DenseP P, float* __restrict__ out)
{
  __shared__ __align__(16) _Float16 Ash[8][16][128];  // per-wave act (wide)
  __shared__ __align__(16) _Float16 Bsh[8][16][64];   // per-wave act (narrow)

  const int tid  = threadIdx.x;
  const int lane = tid & 31;
  const int w    = tid >> 5;
  const int s0   = blockIdx.x * 128 + w * 16;

  _Float16* a  = &Ash[w][0][0];
  _Float16* bm = &Bsh[w][0][0];

  // stage poster features into A cols 32..63 (kept for fusion concat)
  for (int idx = lane; idx < 16 * 32; idx += 32) {
    int mm = idx >> 5, nn = idx & 31;
    a[mm * 128 + 32 + nn] = (_Float16)pf[(size_t)(s0 + mm) * 32 + nn];
  }
  // v = Wv*pf + bv                -> B cols 0..31
  linear16<32, 0>(a, 128, 32, dfr, FB_WV, P.bv, bm, 64, 0, lane);
  // attn_out = Wo*v + bo          -> A cols 0..31   ([attn | pf] now in A)
  linear16<32, 0>(bm, 64, 0, dfr, FB_WO, P.bo, a, 128, 0, lane);
  // fusion1 = relu(W1*[attn|pf])  -> B cols 0..31   (K = 64)
  linear16<64, 1>(a, 128, 0, dfr, FB_F1, P.f1b, bm, 64, 0, lane);
  // fusion2 = relu(W2*fusion1)    -> A cols 0..31
  linear16<32, 1>(bm, 64, 0, dfr, FB_F2, P.f2b, a, 128, 0, lane);
  // movie_feat = tanh(Wmf*fusion2)-> B cols 32..63
  linear16<32, 2>(a, 128, 0, dfr, FB_MF, P.mfb, bm, 64, 32, lane);

  // ---- user tower: gather embeddings straight from the tables ----
  for (int idx = lane; idx < 16 * 32; idx += 32) {   // user_emb -> A cols 0..31
    int mm = idx >> 5, nn = idx & 31;
    a[mm * 128 + nn] = (_Float16)P.ue[(size_t)P.uid[s0 + mm] * 32 + nn];
  }
  linear16<32, 1>(a, 128, 0, dfr, FB_UW, P.ub, a, 128, 0, lane);  // in-place

  for (int idx = lane; idx < 16 * 16; idx += 32) {   // gender -> B cols 0..15
    int mm = idx >> 4, nn = idx & 15;
    bm[mm * 64 + nn] = (_Float16)P.ge[(size_t)P.gid[s0 + mm] * 16 + nn];
  }
  linear16<16, 1>(bm, 64, 0, dfr, FB_GW, P.gb, a, 128, 32, lane);

  for (int idx = lane; idx < 16 * 16; idx += 32) {   // age -> B cols 0..15
    int mm = idx >> 4, nn = idx & 15;
    bm[mm * 64 + nn] = (_Float16)P.ae[(size_t)P.aid[s0 + mm] * 16 + nn];
  }
  linear16<16, 1>(bm, 64, 0, dfr, FB_AW, P.ab, a, 128, 64, lane);

  for (int idx = lane; idx < 16 * 16; idx += 32) {   // job -> B cols 0..15
    int mm = idx >> 4, nn = idx & 15;
    bm[mm * 64 + nn] = (_Float16)P.je[(size_t)P.jid[s0 + mm] * 16 + nn];
  }
  linear16<16, 1>(bm, 64, 0, dfr, FB_JW, P.jb, a, 128, 96, lane);

  // user_feat = tanh(Wuf * user_cat[128]) -> B cols 0..31 (4 k-steps)
  linear16<128, 2>(a, 128, 0, dfr, FB_UF, P.ufb, bm, 64, 0, lane);

  // ---- final rowwise dot: out[b] = sum(user_feat * movie_feat) ----
  if (lane < 16) {
    float s = 0.f;
#pragma unroll
    for (int nn = 0; nn < 32; ++nn)
      s += (float)bm[lane * 64 + nn] * (float)bm[lane * 64 + 32 + nn];
    out[s0 + lane] = s;
  }
}

// ---------------------------------------------------------------------------
extern "C" void kernel_launch(void* const* d_in, const int* in_sizes, int n_in,
                              void* d_out, int out_size, void* d_ws, size_t ws_size,
                              hipStream_t stream) {
  (void)in_sizes; (void)out_size; (void)ws_size;
  if (n_in < IN_COUNT) return;

  float*     ws_pf = (float*)d_ws;                              // [B,32]
  _Float16*  frags = (_Float16*)((char*)d_ws + WS_FRAG_BYTES);  // fragments

  DenseP P;
  P.uid = (const int*)d_in[IN_USER_ID];
  P.gid = (const int*)d_in[IN_GENDER];
  P.aid = (const int*)d_in[IN_AGE];
  P.jid = (const int*)d_in[IN_JOB];
  P.ue  = (const float*)d_in[IN_USER_EMB];
  P.ge  = (const float*)d_in[IN_GENDER_EMB];
  P.ae  = (const float*)d_in[IN_AGE_EMB];
  P.je  = (const float*)d_in[IN_JOB_EMB];
  P.wv  = (const float*)d_in[IN_WV];  P.bv  = (const float*)d_in[IN_BV];
  P.wo  = (const float*)d_in[IN_WO];  P.bo  = (const float*)d_in[IN_BO];
  P.f1w = (const float*)d_in[IN_F1W]; P.f1b = (const float*)d_in[IN_F1B];
  P.f2w = (const float*)d_in[IN_F2W]; P.f2b = (const float*)d_in[IN_F2B];
  P.uw  = (const float*)d_in[IN_UW];  P.ub  = (const float*)d_in[IN_UB];
  P.gw  = (const float*)d_in[IN_GW];  P.gb  = (const float*)d_in[IN_GB];
  P.aw  = (const float*)d_in[IN_AW];  P.ab  = (const float*)d_in[IN_AB];
  P.jw  = (const float*)d_in[IN_JW];  P.jb  = (const float*)d_in[IN_JB];
  P.ufw = (const float*)d_in[IN_UFW]; P.ufb = (const float*)d_in[IN_UFB];
  P.mfw = (const float*)d_in[IN_MFW]; P.mfb = (const float*)d_in[IN_MFB];

  // 0) pre-swizzle all weight fragments (single tiny block)
  init_frags_kernel<<<1, 256, 0, stream>>>(
      (const float*)d_in[IN_C1W], (const float*)d_in[IN_C2W], P, frags);

  // 1) poster CNN -> poster_features
  poster_cnn_kernel<<<BATCH, 256, 0, stream>>>(
      (const float*)d_in[IN_POSTER], frags,
      (const float*)d_in[IN_C1B], (const float*)d_in[IN_C2B],
      (const float*)d_in[IN_C3W], (const float*)d_in[IN_C3B],
      (const float*)d_in[IN_PFW], (const float*)d_in[IN_PFB],
      ws_pf);

  // 2) dense towers + final dot
  dense_wmma_kernel<<<BATCH / 128, 256, 0, stream>>>(
      ws_pf, frags + FR_DENSE, P, (float*)d_out);
}
